// DepthAwareBokehDFN_33131377721513
// MI455X (gfx1250) — matrix operand order repaired and checked by reading, not compiled, not run on hardware
//
#include <hip/hip_runtime.h>
#include <hip/hip_bf16.h>

// ---------------------------------------------------------------------------
// DepthAwareBokehDFN for MI455X (gfx1250, wave32, WMMA 16x16x32 bf16 + TDM)
//
// Pipeline:
//   1) pack_weights: (O,I,3,3) f32 -> bf16 B-fragments in exact WMMA lane order
//   2) conv1: direct 4->64 conv (K=36, too small for WMMA), relu, bf16 NHWC out
//   3) conv_wmma<4,relu,bf16>:  64->64 implicit GEMM; per-tap weights staged
//      into LDS by the Tensor Data Mover (tensor_load_to_lds, TENSORcnt),
//      double-buffered so tap t+1 DMA overlaps tap t WMMA.
//   4) conv_wmma<6,none,f32>:   64->96(=81 padded) implicit GEMM -> logits
//   5) bokeh: softmax over 81 taps + 9x9 dynamic-filter gather of rgb
// ---------------------------------------------------------------------------

typedef __bf16 bf16_t;
typedef __attribute__((ext_vector_type(16))) __bf16   v16bf;
typedef __attribute__((ext_vector_type(8)))  float    v8f;
typedef __attribute__((ext_vector_type(4)))  unsigned int u32x4;
typedef __attribute__((ext_vector_type(8)))  int      i32x8;
typedef __attribute__((ext_vector_type(4)))  int      i32x4;

union Frag {                 // one WMMA 16-bit A or B operand: 8 VGPRs / lane
    u32x4 q[2];
    v16bf v;
};

#define Hc 384
#define Wc 384
#define Bc 4
#define CIN 64

// ---------------------------------------------------------------------------
// TDM: 1-row 2D tile load, Global -> LDS. n8 = payload in 8-byte units.
// D# built per CDNA5 ISA 8.3/8.4:
//   group0: count=1 | lds_addr[63:32] | global_addr[120:64] | type=2 [127:126]
//   group1: data_size=3 (8B) [17:16]; tensor_dim0[79:48]=n8; tensor_dim1=1;
//           tile_dim0[127:112]=n8; tile_dim1[143:128]=1; dim0_stride=n8
//   groups 2/3/4: zero (tile_dim2/3 unused -> <=2D tensor)
// ---------------------------------------------------------------------------
__device__ __forceinline__ void tdm_stage(const void* gsrc, void* ldst, int n8)
{
    const unsigned lds_addr = (unsigned)(size_t)ldst;        // LDS byte offset
    const unsigned long long ga = (unsigned long long)(size_t)gsrc;

    u32x4 g0;
    g0.x = 1u;                                               // count=1, user D#
    g0.y = lds_addr;
    g0.z = (unsigned)ga;                                     // global_addr[31:0]
    g0.w = (unsigned)((ga >> 32) & 0x01FFFFFFu) | (2u << 30);// [56:32] | type=2

    i32x8 g1 = {};
    g1[0] = (3 << 16);                                       // data_size = 8B
    g1[1] = (int)(((unsigned)n8 & 0xFFFFu) << 16);           // tensor_dim0 lo
    g1[2] = (int)((((unsigned)n8 >> 16) & 0xFFFFu) | (1u << 16)); // dim0 hi | tensor_dim1=1
    g1[3] = (int)(((unsigned)n8 & 0xFFFFu) << 16);           // tile_dim0 = n8
    g1[4] = 1;                                               // tile_dim1 = 1
    g1[5] = n8;                                              // tensor_dim0_stride
    i32x4 g2 = {};
    i32x4 g3 = {};
    i32x8 g4 = {};
    __builtin_amdgcn_tensor_load_to_lds(g0, g1, g2, g3, g4, 0);
}

// ---------------------------------------------------------------------------
// Compile-time nt loop so the WMMA reuse_a hint is a constant expression.
// A operand is identical across the NT consecutive WMMAs of one k-step:
// set reuse_a on all but the last (ISA 7.12 operand-reuse rule).
// ---------------------------------------------------------------------------
template<int NT, int NTI>
__device__ __forceinline__ void wmma_ntiles(const v16bf& a,
                                            const bf16_t* __restrict__ wbase,
                                            int lane, v8f* acc)
{
    if constexpr (NTI < NT) {
        Frag bf;
        const bf16_t* wsrc = wbase + ((size_t)NTI * 32 + lane) * 16;
        bf.q[0] = *(const u32x4*)(wsrc);       // ds_load_b128
        bf.q[1] = *(const u32x4*)(wsrc + 8);   // ds_load_b128
        acc[NTI] = __builtin_amdgcn_wmma_f32_16x16x32_bf16(
            /*neg_a=*/false, a, /*neg_b=*/false, bf.v,
            /*c_mod=*/(short)0, acc[NTI],
            /*reuse_a=*/(NTI < NT - 1), /*reuse_b=*/false);
        wmma_ntiles<NT, NTI + 1>(a, wbase, lane, acc);
    }
}

// ---------------------------------------------------------------------------
// Pack (O,I,3,3) f32 weights into bf16 WMMA B-fragments:
//   layout: [tap(9)][kstep(2)][ntile(NT)][lane(32)][elem(16)]
//   B 32x16 (KxN) bf16 layout: lane L -> N = L%16;
//     lanes 0-15 hold K=0..7,16..23; lanes 16-31 hold K=8..15,24..31 (matches A).
//   ci = kstep*32 + (e<8 ? (L/16)*8 + e : 16 + (L/16)*8 + e-8) ; co = ntile*16 + (L%16)
// ---------------------------------------------------------------------------
__global__ __launch_bounds__(256)
void pack_weights_kernel(const float* __restrict__ w, bf16_t* __restrict__ wpack,
                         int nt_count, int co_real)
{
    const int idx = blockIdx.x * 256 + threadIdx.x;
    const int total = 9 * 2 * nt_count * 512;
    if (idx >= total) return;

    const int e    = idx & 15;
    const int lane = (idx >> 4) & 31;
    const int nt   = (idx >> 9) % nt_count;
    const int ks   = (idx / (512 * nt_count)) % 2;
    const int tap  =  idx / (1024 * nt_count);

    const int v  = e >> 1;
    const int j  = e & 1;
    const int ci = ks * 32 + (lane >> 4) * 8 + ((2 * v + j) & 7) + ((2 * v + j) >> 3) * 16;   // 0..63
    const int co = nt * 16 + (lane & 15);
    const int kh = tap / 3, kw = tap % 3;

    float val = 0.0f;
    if (co < co_real)
        val = w[(((size_t)co * CIN + ci) * 3 + kh) * 3 + kw];
    wpack[idx] = (bf16_t)val;
}

// ---------------------------------------------------------------------------
// conv1: concat(rgb,depth) 4ch -> 64ch, 3x3 SAME, relu, bf16 NHWC output.
// Weights staged in LDS (2304 f32 = 9.2 KB); one thread per pixel.
// ---------------------------------------------------------------------------
__global__ __launch_bounds__(256)
void conv1_kernel(const float* __restrict__ rgb, const float* __restrict__ depth,
                  const float* __restrict__ w1, const float* __restrict__ b1,
                  bf16_t* __restrict__ h1)
{
    __shared__ float wls[64 * 36];
    __shared__ float bls[64];
    for (int i = threadIdx.x; i < 64 * 36; i += 256) wls[i] = w1[i];
    if (threadIdx.x < 64) bls[threadIdx.x] = b1[threadIdx.x];
    __syncthreads();

    const int pix = blockIdx.x * 256 + threadIdx.x;
    const int b = pix / (Hc * Wc);
    const int r = pix % (Hc * Wc);
    const int y = r / Wc, x = r % Wc;

    float patch[36];
    #pragma unroll
    for (int c = 0; c < 4; ++c) {
        const float* src = (c < 3) ? (rgb + ((size_t)(b * 3 + c) * Hc) * Wc)
                                   : (depth + ((size_t)b * Hc) * Wc);
        #pragma unroll
        for (int t = 0; t < 9; ++t) {
            const int py = y + t / 3 - 1;
            const int px = x + t % 3 - 1;
            const bool valid = (py >= 0) & (py < Hc) & (px >= 0) & (px < Wc);
            patch[c * 9 + t] = valid ? src[(size_t)py * Wc + px] : 0.0f;
        }
    }

    bf16_t* dst = h1 + (size_t)pix * 64;
    #pragma unroll 4
    for (int co = 0; co < 64; ++co) {
        float s = bls[co];
        #pragma unroll
        for (int i = 0; i < 36; ++i) s += patch[i] * wls[co * 36 + i];
        dst[co] = (bf16_t)fmaxf(s, 0.0f);
    }
}

// ---------------------------------------------------------------------------
// Implicit-GEMM 3x3 conv via WMMA bf16 (wave32) + TDM weight staging.
//   in:    NHWC bf16, 64 channels
//   wpack: packed B-fragments (see pack_weights_kernel)
//   Each wave: M = 16 contiguous x-pixels on one row, N = NT*16 channels,
//              K = 9 taps * 64 ch = 576 -> 18 WMMA k-steps per N-tile.
//   A 16x32 bf16 layout: lane L -> M = L%16;
//     lanes 0-15: VGPR0-3 = K 0..7 (pairs), VGPR4-7 = K 16..23
//     lanes 16-31: VGPR0-3 = K 8..15,       VGPR4-7 = K 24..31
//   => per k-step lane loads two contiguous 16B chunks at channel offsets
//      (half*8) and (half*8+16) -- NHWC makes these plain b128 loads.
//   Per-tap B fragments (NT*1024 bf16) are DMA'd into a double-buffered LDS
//   tile by wave 0 via tensor_load_to_lds; all 8 waves consume via ds_load.
// ---------------------------------------------------------------------------
template<int NT, bool RELU, bool OUTF32>
__global__ __launch_bounds__(256)
void conv_wmma_kernel(const bf16_t* __restrict__ in,
                      const bf16_t* __restrict__ wpack,
                      const float* __restrict__ bias, int nco_real,
                      bf16_t* __restrict__ out_bf, float* __restrict__ out_f32)
{
    constexpr int TAP_ELEMS = NT * 1024;           // bf16 per tap (2 ksteps * NT * 512)
    constexpr int TAP_N8    = TAP_ELEMS * 2 / 8;   // 8-byte units per tap
    __shared__ bf16_t wlds[2][TAP_ELEMS];          // double buffer (16/24 KB)

    const int lane = threadIdx.x & 31;
    const int wave = threadIdx.x >> 5;
    const int strip = blockIdx.x * 8 + wave;      // 16-pixel strip id
    const int spr = Wc / 16;                      // strips per row = 24
    const int b  = strip / (Hc * spr);
    const int r  = strip % (Hc * spr);
    const int y  = r / spr;
    const int xs = (r % spr) * 16;

    const int m    = lane & 15;
    const int half = lane >> 4;

    v8f acc[NT] = {};

    // Prologue: DMA tap 0's fragments into buffer 0 (one TDM op per block).
    if (wave == 0)
        tdm_stage(wpack, &wlds[0][0], TAP_N8);

    #pragma unroll
    for (int tap = 0; tap < 9; ++tap) {
        const int buf = tap & 1;

        // Issue next tap's DMA into the other buffer (safe: it was last read
        // at tap-1 and the trailing barrier of that iteration has passed).
        if (wave == 0) {
            if (tap < 8)
                tdm_stage(wpack + (size_t)(tap + 1) * TAP_ELEMS,
                          &wlds[buf ^ 1][0], TAP_N8);
            // Wait for tap's DMA: <=1 outstanding (= the one just issued).
            if (tap < 8) __builtin_amdgcn_s_wait_tensorcnt(1);
            else         __builtin_amdgcn_s_wait_tensorcnt(0);
        }
        __syncthreads();                           // publish wlds[buf]

        const int dy = tap / 3 - 1;
        const int dx = tap % 3 - 1;
        const int py = y + dy;
        const int px = xs + m + dx;
        const bool valid = (py >= 0) & (py < Hc) & (px >= 0) & (px < Wc);
        const int pyc = py < 0 ? 0 : (py >= Hc ? Hc - 1 : py);
        const int pxc = px < 0 ? 0 : (px >= Wc ? Wc - 1 : px);
        const bf16_t* src = in + (((size_t)b * Hc + pyc) * Wc + pxc) * CIN;

        // all 64 channels of this tap's A column for this lane: 4 x b128
        u32x4 q0 = {}, q1 = {}, q2 = {}, q3 = {};
        if (valid) {
            q0 = *(const u32x4*)(src + half * 8);
            q1 = *(const u32x4*)(src + half * 8 + 16);
            q2 = *(const u32x4*)(src + half * 8 + 32);
            q3 = *(const u32x4*)(src + half * 8 + 48);
        }

        // warm L2 for the DMA source two taps ahead (global_prefetch_b8)
        if (tap < 7)
            __builtin_prefetch(wpack + (size_t)(tap + 2) * TAP_ELEMS, 0, 1);

        #pragma unroll
        for (int ks = 0; ks < 2; ++ks) {
            Frag a;
            a.q[0] = ks ? q2 : q0;
            a.q[1] = ks ? q3 : q1;
            const bf16_t* wbase = &wlds[buf][(size_t)ks * NT * 512];
            wmma_ntiles<NT, 0>(a.v, wbase, lane, acc);
        }
        __syncthreads();                           // done reading wlds[buf]
    }

    // Epilogue. C/D layout: lane L -> N = L%16, M = (L/16)*8 + v.
    #pragma unroll
    for (int nt = 0; nt < NT; ++nt) {
        const int co = nt * 16 + m;
        const float bv = (co < nco_real) ? bias[co] : 0.0f;
        #pragma unroll
        for (int v = 0; v < 8; ++v) {
            float val = acc[nt][v] + bv;
            if (RELU) val = fmaxf(val, 0.0f);
            const int px2 = xs + half * 8 + v;
            const size_t oidx = (((size_t)b * Hc + y) * Wc + px2) * (NT * 16) + co;
            if (OUTF32) out_f32[oidx] = val;
            else        out_bf[oidx]  = (bf16_t)val;
        }
    }
}

// ---------------------------------------------------------------------------
// Softmax over 81 taps + 9x9 dynamic filter applied to rgb. One thread/pixel.
// logits: NHWC, stride 96 (channels 81..95 are padding, ignored).
// ---------------------------------------------------------------------------
__global__ __launch_bounds__(256)
void bokeh_kernel(const float* __restrict__ logits, const float* __restrict__ rgb,
                  float* __restrict__ out)
{
    const int pix = blockIdx.x * 256 + threadIdx.x;
    const int b = pix / (Hc * Wc);
    const int r = pix % (Hc * Wc);
    const int y = r / Wc, x = r % Wc;

    const float* lg = logits + (size_t)pix * 96;

    float mx = -3.4e38f;
    #pragma unroll
    for (int t = 0; t < 81; ++t) mx = fmaxf(mx, lg[t]);
    float sum = 0.0f;
    #pragma unroll
    for (int t = 0; t < 81; ++t) sum += __expf(lg[t] - mx);
    const float inv = 1.0f / sum;

    float a0 = 0.f, a1 = 0.f, a2 = 0.f;
    #pragma unroll
    for (int dy = 0; dy < 9; ++dy) {
        const int py = y + dy - 4;
        if (py < 0 || py >= Hc) continue;
        #pragma unroll
        for (int dx = 0; dx < 9; ++dx) {
            const int px = x + dx - 4;
            if (px < 0 || px >= Wc) continue;
            const float f = __expf(lg[dy * 9 + dx] - mx) * inv;
            const size_t base = ((size_t)b * 3 * Hc) * Wc + (size_t)py * Wc + px;
            a0 += f * rgb[base];
            a1 += f * rgb[base + (size_t)Hc * Wc];
            a2 += f * rgb[base + 2 * (size_t)Hc * Wc];
        }
    }
    const size_t obase = ((size_t)b * 3 * Hc) * Wc + (size_t)y * Wc + x;
    out[obase]                       = a0;
    out[obase + (size_t)Hc * Wc]     = a1;
    out[obase + 2 * (size_t)Hc * Wc] = a2;
}

// ---------------------------------------------------------------------------
extern "C" void kernel_launch(void* const* d_in, const int* in_sizes, int n_in,
                              void* d_out, int out_size, void* d_ws, size_t ws_size,
                              hipStream_t stream)
{
    const float* rgb   = (const float*)d_in[0];
    const float* depth = (const float*)d_in[1];
    const float* w1 = (const float*)d_in[2];
    const float* b1 = (const float*)d_in[3];
    const float* w2 = (const float*)d_in[4];
    const float* b2 = (const float*)d_in[5];
    const float* w3 = (const float*)d_in[6];
    const float* b3 = (const float*)d_in[7];
    float* out = (float*)d_out;

    const size_t BHW = (size_t)Bc * Hc * Wc;                 // 589824
    char* ws = (char*)d_ws;

    // logits (f32, stride 96) aliases h1 (bf16) -- h1 is dead before conv3 runs.
    float*  logits = (float*)ws;
    bf16_t* h1     = (bf16_t*)ws;
    ws += BHW * 96 * sizeof(float);                          // 226.5 MB
    bf16_t* h2 = (bf16_t*)ws;  ws += BHW * 64 * sizeof(bf16_t); // 75.5 MB
    bf16_t* pw2 = (bf16_t*)ws; ws += (size_t)9 * 2 * 4 * 512 * sizeof(bf16_t);
    bf16_t* pw3 = (bf16_t*)ws;

    pack_weights_kernel<<<(9 * 2 * 4 * 512 + 255) / 256, 256, 0, stream>>>(w2, pw2, 4, 64);
    pack_weights_kernel<<<(9 * 2 * 6 * 512 + 255) / 256, 256, 0, stream>>>(w3, pw3, 6, 81);

    conv1_kernel<<<(int)(BHW / 256), 256, 0, stream>>>(rgb, depth, w1, b1, h1);

    const int strips = Bc * Hc * (Wc / 16);                  // 36864
    conv_wmma_kernel<4, true,  false><<<strips / 8, 256, 0, stream>>>(
        h1, pw2, b2, 64, h2, nullptr);
    conv_wmma_kernel<6, false, true><<<strips / 8, 256, 0, stream>>>(
        h2, pw3, b3, 81, nullptr, logits);

    bokeh_kernel<<<(int)(BHW / 256), 256, 0, stream>>>(logits, rgb, out);
}
